// DoomAgent_3599182594583
// MI455X (gfx1250) — matrix operand order repaired and checked by reading, not compiled, not run on hardware
//
#include <hip/hip_runtime.h>
#include <hip/hip_bf16.h>
#include <math.h>

typedef __bf16 bf16_t;
typedef bf16_t v16bf __attribute__((ext_vector_type(16)));
typedef bf16_t v8bf  __attribute__((ext_vector_type(8)));
typedef bf16_t v4bf  __attribute__((ext_vector_type(4)));
typedef float  v8f   __attribute__((ext_vector_type(8)));

#define ACT_NONE     0
#define ACT_RELU     1
#define ACT_LEAKY02  2
#define ACT_LEAKY001 3

#define AM_DIRECT 0   // A bf16 row-major, 16B-aligned fragments
#define AM_CONV   1   // implicit im2col, scalar gathers via LDS offset table
#define AM_CONVW8 2   // implicit im2col, KW==8: row-contiguous v4bf gathers

#define OM_F32  0
#define OM_BF16 1
#define OM_NCHW 2

template<int ACT>
__device__ __forceinline__ float apply_act(float x) {
  if (ACT == ACT_RELU)     return x > 0.f ? x : 0.f;
  if (ACT == ACT_LEAKY02)  return x > 0.f ? x : 0.2f * x;
  if (ACT == ACT_LEAKY001) return x > 0.f ? x : 0.01f * x;
  return x;
}

// C[M,N] = act(A[M,K] @ B[N,K]^T + bias)
// 256 threads = 8 waves; each wave owns a 16x32 output tile (two 16x16 WMMA
// accumulators sharing one A fragment). Block tile = 128(M) x 32(N).
// Conv modes gather A implicitly from NCHW bf16; k->offset decode is done once
// per block into an LDS table (no divisions in the K loop).
template<int AMODE, int ACT, int OMODE>
__global__ __launch_bounds__(256)
void wmma_gemm(const bf16_t* __restrict__ A, const bf16_t* __restrict__ B,
               const float* __restrict__ bias, void* __restrict__ Cout,
               int M, int N, int K, int lda, int ldout, int n_valid,
               int IC, int IH, int IW, int KH, int KW, int S, int OH, int OW)
{
  const int lane = threadIdx.x & 31;
  const int wave = threadIdx.x >> 5;
  const int mt = blockIdx.y * 8 + wave;   // 16-row M tile
  const int nt = blockIdx.x;              // 32-col N tile

  const int ohow = OH * OW;
  const int khkw = KH * KW;

  constexpr bool isconv = (AMODE != AM_DIRECT);
  __shared__ int koff[isconv ? 640 : 1];
  if constexpr (isconv) {
    // one-time k -> (ic*IH + ky)*IW + kx table (divisions hoisted out of loop)
    for (int k = threadIdx.x; k < K; k += 256) {
      int ic = k / khkw;
      int r  = k - ic * khkw;
      int ky = r / KW;
      int kx = r - ky * KW;
      koff[k] = (ic * IH + ky) * IW + kx;
    }
    __syncthreads();   // all 256 threads reach this before any wave can exit
  }

  if (mt * 16 >= M) return;  // uniform per-wave exit; EXEC stays all-ones

  const int ahalf = lane >> 4;
  const int m_a   = mt * 16 + (lane & 15);   // A row this lane feeds
  const int nb0   = nt * 32 + (lane & 15);   // B column (first 16-wide tile)

  size_t abase = 0;
  if constexpr (isconv) {
    int img = m_a / ohow;
    int r   = m_a - img * ohow;
    int oy  = r / OW;
    int ox  = r - oy * OW;
    abase = (size_t)img * IC * IH * IW + (size_t)(oy * S) * IW + (size_t)(ox * S);
  }

  v8f acc0 = {}, acc1 = {};
  for (int k0 = 0; k0 < K; k0 += 32) {
    const int kb = k0 + ahalf * 8;
    union { v16bf v; v8bf h[2]; v4bf q[4]; unsigned short u[16]; } af;
    if constexpr (AMODE == AM_DIRECT) {
      const bf16_t* ap = A + (size_t)m_a * lda + kb;
      af.h[0] = *(const v8bf*)(ap);        // k = kb .. kb+7
      af.h[1] = *(const v8bf*)(ap + 16);   // k = kb+16 .. kb+23
    } else if constexpr (AMODE == AM_CONVW8) {
      // KW==8, kb%8==0: each 8-group is one contiguous patch row (8B aligned)
      const bf16_t* ap0 = A + abase + koff[kb];
      const bf16_t* ap1 = A + abase + koff[kb + 16];
      af.q[0] = *(const v4bf*)(ap0);
      af.q[1] = *(const v4bf*)(ap0 + 4);
      af.q[2] = *(const v4bf*)(ap1);
      af.q[3] = *(const v4bf*)(ap1 + 4);
    } else {
      // two-phase: issue all 16 independent loads, then pack (waits coalesce)
      unsigned short tmp[16];
      #pragma unroll
      for (int e = 0; e < 16; ++e) {
        int k = kb + ((e < 8) ? e : (8 + e));
        tmp[e] = *(const unsigned short*)(A + abase + koff[k]);
      }
      #pragma unroll
      for (int e = 0; e < 16; ++e) af.u[e] = tmp[e];
    }
    const bf16_t* bp = B + (size_t)nb0 * K + kb;
    union { v16bf v; v8bf h[2]; } ub0, ub1;
    ub0.h[0] = *(const v8bf*)(bp);
    ub0.h[1] = *(const v8bf*)(bp + 16);
    ub1.h[0] = *(const v8bf*)(bp + (size_t)16 * K);
    ub1.h[1] = *(const v8bf*)(bp + (size_t)16 * K + 16);
    v16bf bf0 = ub0.v;
    v16bf bf1 = ub1.v;
    acc0 = __builtin_amdgcn_wmma_f32_16x16x32_bf16(false, af.v, false, bf0,
                                                   (short)0, acc0, false, false);
    acc1 = __builtin_amdgcn_wmma_f32_16x16x32_bf16(false, af.v, false, bf1,
                                                   (short)0, acc1, false, false);
  }

  // epilogue: lane holds C[mbase+v][n] for both 16-wide halves
  const int mbase = mt * 16 + ahalf * 8;
  #pragma unroll
  for (int half = 0; half < 2; ++half) {
    const v8f acc = half ? acc1 : acc0;
    const int n = nt * 32 + half * 16 + (lane & 15);
    const bool nok = (n < n_valid);
    const float bv = (bias != nullptr && nok) ? bias[n] : 0.f;
    #pragma unroll
    for (int v = 0; v < 8; ++v) {
      const int m = mbase + v;
      float x = apply_act<ACT>(acc[v] + bv);
      if constexpr (OMODE == OM_F32) {
        if (nok) ((float*)Cout)[(size_t)m * ldout + n] = x;
      } else if constexpr (OMODE == OM_BF16) {
        ((bf16_t*)Cout)[(size_t)m * ldout + n] = (bf16_t)x;
      } else { // OM_NCHW: m -> (img, oy, ox); channel = n (OC == N)
        int ni = m / ohow;
        int r  = m - ni * ohow;
        int oy = r / OW;
        int ox = r - oy * OW;
        ((bf16_t*)Cout)[(((size_t)ni * N + n) * OH + oy) * OW + ox] = (bf16_t)x;
      }
    }
  }
}

// ---- small elementwise kernels --------------------------------------------

__global__ void k_f32_to_bf16(const float* __restrict__ s, bf16_t* __restrict__ d,
                              int nsrc, int ndst) {
  int i = blockIdx.x * blockDim.x + threadIdx.x;
  if (i < ndst) d[i] = (i < nsrc) ? (bf16_t)s[i] : (bf16_t)0.f;
}

__global__ void k_add_bias2(const float* __restrict__ a, const float* __restrict__ b,
                            float* __restrict__ d, int n) {
  int i = blockIdx.x * blockDim.x + threadIdx.x;
  if (i < n) d[i] = a[i] + b[i];
}

__global__ void k_init_state(const float* __restrict__ h0, const float* __restrict__ c0,
                             float* __restrict__ h, float* __restrict__ c) {
  int i = blockIdx.x * blockDim.x + threadIdx.x;
  if (i < 16384) { h[i] = h0[i]; c[i] = c0[i]; }
}

__global__ void k_lstm_mask(float* __restrict__ h, float* __restrict__ c,
                            bf16_t* __restrict__ hb, const int* __restrict__ done, int t) {
  int i = blockIdx.x * blockDim.x + threadIdx.x;
  if (i >= 16384) return;
  int b = i >> 9;
  float m = 1.f - (float)done[t * 32 + b];
  float hv = h[i] * m;
  h[i] = hv;
  c[i] = c[i] * m;
  hb[i] = (bf16_t)hv;
}

__device__ __forceinline__ float sigm(float x) { return 1.f / (1.f + expf(-x)); }

// gates: GX already contains x@Wih^T + (bih+bhh); GH contains h@Whh^T
__global__ void k_lstm_cell(const float* __restrict__ GX, const float* __restrict__ GH,
                            float* __restrict__ h, float* __restrict__ c,
                            bf16_t* __restrict__ hb, bf16_t* __restrict__ nhb, int t) {
  int i = blockIdx.x * blockDim.x + threadIdx.x;
  if (i >= 16384) return;
  int b = i >> 9, n = i & 511;
  size_t gx = (size_t)(t * 32 + b) * 2048;
  size_t gh = (size_t)b * 2048;
  float gi = GX[gx + n]        + GH[gh + n];
  float gf = GX[gx + 512 + n]  + GH[gh + 512 + n];
  float gg = GX[gx + 1024 + n] + GH[gh + 1024 + n];
  float go = GX[gx + 1536 + n] + GH[gh + 1536 + n];
  float cs = sigm(gf) * c[i] + sigm(gi) * tanhf(gg);
  float hs = sigm(go) * tanhf(cs);
  c[i] = cs;
  h[i] = hs;
  hb[i] = (bf16_t)hs;
  nhb[(size_t)(t * 32 + b) * 512 + n] = (bf16_t)hs;
}

__global__ void k_copy_state_out(const float* __restrict__ h, const float* __restrict__ c,
                                 float* __restrict__ out) {
  int i = blockIdx.x * blockDim.x + threadIdx.x;
  if (i < 16384) out[i] = h[i];
  else if (i < 32768) out[i] = c[i - 16384];
}

// ---- launcher --------------------------------------------------------------

extern "C" void kernel_launch(void* const* d_in, const int* in_sizes, int n_in,
                              void* d_out, int out_size, void* d_ws, size_t ws_size,
                              hipStream_t stream) {
  (void)in_sizes; (void)n_in; (void)out_size; (void)ws_size;

  const float* x    = (const float*)d_in[0];
  const int*   done = (const int*)d_in[1];
  const float* h0   = (const float*)d_in[2];
  const float* c0   = (const float*)d_in[3];
  const float* w1   = (const float*)d_in[4];
  const float* b1   = (const float*)d_in[5];
  const float* w2   = (const float*)d_in[6];
  const float* b2   = (const float*)d_in[7];
  const float* w3   = (const float*)d_in[8];
  const float* b3   = (const float*)d_in[9];
  const float* fcW  = (const float*)d_in[10];
  const float* fcb  = (const float*)d_in[11];
  const float* Wih  = (const float*)d_in[12];
  const float* Whh  = (const float*)d_in[13];
  const float* bih  = (const float*)d_in[14];
  const float* bhh  = (const float*)d_in[15];
  const float* A1   = (const float*)d_in[16];
  const float* Ab1  = (const float*)d_in[17];
  const float* A2   = (const float*)d_in[18];
  const float* Ab2  = (const float*)d_in[19];
  const float* C1m  = (const float*)d_in[20];
  const float* Cb1  = (const float*)d_in[21];
  const float* C2m  = (const float*)d_in[22];
  const float* Cb2  = (const float*)d_in[23];

  const int NX = 512 * 3 * 120 * 160;      // 29,491,200
  const int M1 = 512 * 29 * 39;            // 579,072
  const int M2 = 512 * 13 * 18;            // 119,808
  const int M3 = 512 * 11 * 16;            // 90,112

  char* ws = (char*)d_ws;
  size_t off = 0;
  auto alloc = [&](size_t bytes) -> void* {
    void* p = ws + off;
    off = (off + bytes + 255) & ~(size_t)255;
    return p;
  };
  bf16_t* xb    = (bf16_t*)alloc((size_t)NX * 2);
  bf16_t* w1b   = (bf16_t*)alloc(6144 * 2);
  bf16_t* w2b   = (bf16_t*)alloc(32768 * 2);
  bf16_t* w3b   = (bf16_t*)alloc(36864 * 2);
  bf16_t* fcWb  = (bf16_t*)alloc((size_t)5767168 * 2);
  bf16_t* Wihb  = (bf16_t*)alloc((size_t)1048576 * 2);
  bf16_t* Whhb  = (bf16_t*)alloc((size_t)1048576 * 2);
  bf16_t* A1b   = (bf16_t*)alloc(262144 * 2);
  bf16_t* A2b   = (bf16_t*)alloc(16384 * 2);    // padded [32,512]
  bf16_t* C1b   = (bf16_t*)alloc(262144 * 2);
  bf16_t* C2b   = (bf16_t*)alloc(16384 * 2);    // padded [32,512]
  bf16_t* O1    = (bf16_t*)alloc((size_t)M1 * 32 * 2);   // NCHW [512,32,29,39]
  bf16_t* O2    = (bf16_t*)alloc((size_t)M2 * 64 * 2);   // NCHW [512,64,13,18]
  bf16_t* O3    = (bf16_t*)alloc((size_t)M3 * 64 * 2);   // NCHW == [512,11264]
  bf16_t* featb = (bf16_t*)alloc(262144 * 2);            // [512,512]
  bf16_t* nhb   = (bf16_t*)alloc(262144 * 2);            // [512,512]
  bf16_t* abuf  = (bf16_t*)alloc(262144 * 2);
  bf16_t* vbuf  = (bf16_t*)alloc(262144 * 2);
  bf16_t* hbb   = (bf16_t*)alloc(16384 * 2);             // [32,512] bf16 h
  float*  GX    = (float*)alloc((size_t)1048576 * 4);    // [512,2048]
  float*  GH    = (float*)alloc((size_t)65536 * 4);      // [32,2048]
  float*  hbuf  = (float*)alloc(16384 * 4);
  float*  cbuf  = (float*)alloc(16384 * 4);
  float*  bsum  = (float*)alloc(2048 * 4);               // bih + bhh

  const dim3 blk(256);
  auto cvt = [&](const float* s, bf16_t* d, int nsrc, int ndst) {
    k_f32_to_bf16<<<dim3((ndst + 255) / 256), blk, 0, stream>>>(s, d, nsrc, ndst);
  };

  // fp32 -> bf16 conversions; A2/C2 zero-padded to 32 output rows
  cvt(x,   xb,   NX, NX);
  cvt(w1,  w1b,  6144, 6144);
  cvt(w2,  w2b,  32768, 32768);
  cvt(w3,  w3b,  36864, 36864);
  cvt(fcW, fcWb, 5767168, 5767168);
  cvt(Wih, Wihb, 1048576, 1048576);
  cvt(Whh, Whhb, 1048576, 1048576);
  cvt(A1,  A1b,  262144, 262144);
  cvt(A2,  A2b,  4096, 16384);
  cvt(C1m, C1b,  262144, 262144);
  cvt(C2m, C2b,  512, 16384);
  k_add_bias2<<<dim3(8), blk, 0, stream>>>(bih, bhh, bsum, 2048);

  // conv1: [M1,32] = im2col(x) @ w1^T, K=192 (KW=8 -> vector row gathers)
  wmma_gemm<AM_CONVW8, ACT_RELU, OM_NCHW><<<dim3(1, 4524), blk, 0, stream>>>(
      xb, w1b, b1, O1, M1, 32, 192, 0, 0, 32,
      3, 120, 160, 8, 8, 4, 29, 39);
  // conv2: [M2,64], K=512
  wmma_gemm<AM_CONV, ACT_RELU, OM_NCHW><<<dim3(2, 936), blk, 0, stream>>>(
      O1, w2b, b2, O2, M2, 64, 512, 0, 0, 64,
      32, 29, 39, 4, 4, 2, 13, 18);
  // conv3: [M3,64], K=576
  wmma_gemm<AM_CONV, ACT_RELU, OM_NCHW><<<dim3(2, 704), blk, 0, stream>>>(
      O2, w3b, b3, O3, M3, 64, 576, 0, 0, 64,
      64, 13, 18, 3, 3, 1, 11, 16);
  // FC: feat[512,512] = O3[512,11264] @ fcW^T, leaky 0.2
  wmma_gemm<AM_DIRECT, ACT_LEAKY02, OM_BF16><<<dim3(16, 4), blk, 0, stream>>>(
      O3, fcWb, fcb, featb, 512, 512, 11264, 11264, 512, 512,
      0, 0, 0, 0, 0, 0, 1, 1);
  // GX[512,2048] = feat @ Wih^T + (bih+bhh)
  wmma_gemm<AM_DIRECT, ACT_NONE, OM_F32><<<dim3(64, 4), blk, 0, stream>>>(
      featb, Wihb, bsum, GX, 512, 2048, 512, 512, 2048, 2048,
      0, 0, 0, 0, 0, 0, 1, 1);

  // recurrent scan
  k_init_state<<<dim3(64), blk, 0, stream>>>(h0, c0, hbuf, cbuf);
  for (int t = 0; t < 16; ++t) {
    k_lstm_mask<<<dim3(64), blk, 0, stream>>>(hbuf, cbuf, hbb, done, t);
    wmma_gemm<AM_DIRECT, ACT_NONE, OM_F32><<<dim3(64, 1), blk, 0, stream>>>(
        hbb, Whhb, nullptr, GH, 32, 2048, 512, 512, 2048, 2048,
        0, 0, 0, 0, 0, 0, 1, 1);
    k_lstm_cell<<<dim3(64), blk, 0, stream>>>(GX, GH, hbuf, cbuf, hbb, nhb, t);
  }

  // actor head
  wmma_gemm<AM_DIRECT, ACT_LEAKY001, OM_BF16><<<dim3(16, 4), blk, 0, stream>>>(
      nhb, A1b, Ab1, abuf, 512, 512, 512, 512, 512, 512,
      0, 0, 0, 0, 0, 0, 1, 1);
  wmma_gemm<AM_DIRECT, ACT_NONE, OM_F32><<<dim3(1, 4), blk, 0, stream>>>(
      abuf, A2b, Ab2, (float*)d_out, 512, 32, 512, 512, 8, 8,
      0, 0, 0, 0, 0, 0, 1, 1);
  // critic head
  wmma_gemm<AM_DIRECT, ACT_LEAKY001, OM_BF16><<<dim3(16, 4), blk, 0, stream>>>(
      nhb, C1b, Cb1, vbuf, 512, 512, 512, 512, 512, 512,
      0, 0, 0, 0, 0, 0, 1, 1);
  wmma_gemm<AM_DIRECT, ACT_NONE, OM_F32><<<dim3(1, 4), blk, 0, stream>>>(
      vbuf, C2b, Cb2, (float*)d_out + 4096, 512, 32, 512, 512, 1, 1,
      0, 0, 0, 0, 0, 0, 1, 1);

  // hT, cT
  k_copy_state_out<<<dim3(128), blk, 0, stream>>>(hbuf, cbuf, (float*)d_out + 4608);
}
